// TransGRU_57372173140520
// MI455X (gfx1250) — compile-verified, matrix operand order checked by reading, not run on hardware
//
#include <hip/hip_runtime.h>
#include <hip/hip_bf16.h>
#include <math.h>

// ---------------------------------------------------------------------------
// TransGRU forward for MI455X (gfx1250, wave32, WMMA).
// Convs lowered to implicit GEMM with v_wmma_f32_16x16x32_f16 (f16 in, f32 acc).
// Weights zero-padded in K to a multiple of 32; conv shapes are template
// parameters so all im2col address math is branch-free constant-folded code.
// ---------------------------------------------------------------------------

typedef __attribute__((ext_vector_type(16))) _Float16 v16h;
typedef __attribute__((ext_vector_type(8)))  _Float16 v8h;
typedef __attribute__((ext_vector_type(8)))  float    v8f;

#define HWPIX 4096   // 64*64 spatial
#define IMGW  64

__device__ __forceinline__ float sigmoidf_(float x) { return 1.f / (1.f + __expf(-x)); }

__device__ __forceinline__ v16h mk16(v8h lo, v8h hi) {
  v16h r;
#pragma unroll
  for (int e = 0; e < 8; ++e) { r[e] = lo[e]; r[e + 8] = hi[e]; }
  return r;
}

// ---------------------------------------------------------------------------
// fp32 -> f16 weight conversion with K zero-padding: dst[Cout][Kpad]
// ---------------------------------------------------------------------------
__global__ void cvt_pad_kernel(const float* __restrict__ src, _Float16* __restrict__ dst,
                               int Cout, int K, int Kpad) {
  int i = blockIdx.x * 256 + threadIdx.x;
  if (i >= Cout * Kpad) return;
  int co = i / Kpad;
  int k  = i - co * Kpad;
  dst[i] = (k < K) ? (_Float16)src[co * K + k] : (_Float16)0.f;
}

// ---------------------------------------------------------------------------
// Implicit-GEMM conv: out[b][co][p] = bias[co] + sum_k W[co][k] * im2col[k][p]
//   k = ci*KHW + kh*3 + kw  (KHW=9 PAD=1, or KHW=1 PAD=0)
//   channels ci<C0 come from src0, else from src1 (fused concat).
// Kpad multiple of 32; padded weight rows are zero, so im2col needs no K
// guard (channel index clamped; product is 0). All shape math compile-time.
// Block = 256 threads (8 waves). Tile: M=64 x N=128. K-step 32.
// ---------------------------------------------------------------------------
template <int C0, int C1, int KHW, int PAD>
__global__ __launch_bounds__(256) void conv_gemm_wmma(
    const _Float16* __restrict__ Wt,   // [Cout][Kpad] f16, zero-padded
    const float* __restrict__ bias,    // [Cout]
    const float* __restrict__ src0, long long s0_bstride,
    const float* __restrict__ src1, long long s1_bstride,
    float* __restrict__ out, int Cout, int Kpad) {
  constexpr int Ctot = C0 + C1;
  const int tid = threadIdx.x;
  const int bb  = blockIdx.z;
  const int M0  = blockIdx.y * 64;
  const int N0  = blockIdx.x * 128;

  __shared__ _Float16 As[64 * 32];    // [m][k]
  __shared__ _Float16 Bs[128 * 32];   // [n][k]

  const int wv    = tid >> 5;
  const int lane  = tid & 31;
  const int wm    = wv >> 2;            // 0..1 (M half)
  const int wn    = wv & 3;             // 0..3 (N quarter)
  const int laneM = lane & 15;
  const int laneK = (lane >> 4) << 3;   // 0 or 8

  v8f acc[2][2] = {};

  // B-tile (im2col) indices: thread -> (row n, k-half)
  const int nB = tid & 127;
  const int hB = tid >> 7;              // 0/1
  const int p  = N0 + nB;
  const int py = p >> 6, px = p & 63;
  const float* s0b = src0 + (size_t)bb * s0_bstride;
  const float* s1b = src1 + (size_t)bb * s1_bstride;

  // A-tile: thread loads 8 contiguous halves (16B) unconditionally
  const int mA = tid >> 2;
  const int kA = (tid & 3) << 3;
  const _Float16* wrow = Wt + (size_t)(M0 + mA) * Kpad + kA;

  for (int k0 = 0; k0 < Kpad; k0 += 32) {
    // ---- stage A (weights): one b128 load + one b128 LDS store
    *(v8h*)(As + mA * 32 + kA) = *(const v8h*)(wrow + k0);

    // ---- stage B (im2col gather), branch-free: clamp + select
    {
      v8h t0, t1;
#pragma unroll
      for (int j = 0; j < 16; ++j) {
        const int kg = k0 + hB * 16 + j;
        int ci, rem;
        if (KHW == 9) { ci = kg / 9; rem = kg - ci * 9; }   // constant-folded
        else          { ci = kg;     rem = 0; }
        if (ci >= Ctot) ci = Ctot - 1;        // padded-K lane: weight is zero
        const int kh = rem / 3;
        const int iy = py + kh - PAD;
        const int ix = px + (rem - kh * 3) - PAD;
        const bool inb = ((unsigned)iy < 64u) && ((unsigned)ix < 64u);
        const int iyc = inb ? iy : 0;
        const int ixc = inb ? ix : 0;
        const float* s = (ci < C0) ? (s0b + (size_t)ci * HWPIX)
                                   : (s1b + (size_t)(ci - C0) * HWPIX);
        float v = s[iyc * IMGW + ixc];        // unconditional, in-bounds
        v = inb ? v : 0.f;
        if (j < 8) t0[j] = (_Float16)v; else t1[j - 8] = (_Float16)v;
      }
      *(v8h*)(Bs + nB * 32 + hB * 16)     = t0;
      *(v8h*)(Bs + nB * 32 + hB * 16 + 8) = t1;
    }
    __syncthreads();

    // ---- fragments + WMMA
    v16h afr[2], bfr[2];
#pragma unroll
    for (int mi = 0; mi < 2; ++mi) {
      int row = wm * 32 + mi * 16 + laneM;
      afr[mi] = mk16(*(const v8h*)(As + row * 32 + laneK),
                     *(const v8h*)(As + row * 32 + laneK + 16));
    }
#pragma unroll
    for (int ni = 0; ni < 2; ++ni) {
      int row = wn * 32 + ni * 16 + laneM;
      bfr[ni] = mk16(*(const v8h*)(Bs + row * 32 + laneK),
                     *(const v8h*)(Bs + row * 32 + laneK + 16));
    }
#pragma unroll
    for (int mi = 0; mi < 2; ++mi)
#pragma unroll
      for (int ni = 0; ni < 2; ++ni)
        acc[mi][ni] = __builtin_amdgcn_wmma_f32_16x16x32_f16(
            false, afr[mi], false, bfr[ni], (short)0, acc[mi][ni], false, false);
    __syncthreads();
  }

  // ---- epilogue: D layout lanes0-15: M=r, lanes16-31: M=r+8
  const int mh = (lane >> 4) * 8;
#pragma unroll
  for (int mi = 0; mi < 2; ++mi) {
#pragma unroll
    for (int ni = 0; ni < 2; ++ni) {
      const int coln = N0 + wn * 32 + ni * 16 + laneM;
#pragma unroll
      for (int r = 0; r < 8; ++r) {
        const int mrow = M0 + wm * 32 + mi * 16 + r + mh;
        out[((size_t)bb * Cout + mrow) * HWPIX + coln] = acc[mi][ni][r] + bias[mrow];
      }
    }
  }
}

// ---------------------------------------------------------------------------
// Gate kernels (elementwise, fp32). Tensors laid out [B=4][64][4096].
// ---------------------------------------------------------------------------
__global__ void gate1_kernel(const float* __restrict__ A, const float* __restrict__ Bm,
                             const float* __restrict__ h, const float* __restrict__ c,
                             const float* __restrict__ m,
                             const float* __restrict__ td_h, const float* __restrict__ td_m,
                             const float* __restrict__ tg,
                             float* __restrict__ r, float* __restrict__ z,
                             float* __restrict__ o, float* __restrict__ xr,
                             float* __restrict__ xz, float* __restrict__ rc,
                             float* __restrict__ xrm) {
  int idx = blockIdx.x * 256 + threadIdx.x;
  if (idx >= 4 * 64 * HWPIX) return;
  int p  = idx & (HWPIX - 1);
  int ch = (idx >> 12) & 63;
  int b  = idx >> 18;
  float g  = sigmoidf_(tg[ch]);
  float dh = __expf(-td_h[ch]);
  float dm = __expf(-td_m[ch]);
  size_t a0 = ((size_t)b * 192 + ch) * HWPIX + p;
  size_t b0 = ((size_t)b * 128 + ch) * HWPIX + p;
  float hv = h[idx], cv = c[idx], mv = m[idx];
  float rv  = sigmoidf_(A[a0] * g + hv * dh);
  float zv  = sigmoidf_(A[a0 + (size_t)64 * HWPIX] * g + hv * dh);
  float ov  = sigmoidf_(A[a0 + (size_t)128 * HWPIX]);
  float xrv = sigmoidf_(Bm[b0] * g + mv * dm);
  float xzv = sigmoidf_(Bm[b0 + (size_t)64 * HWPIX] * g + mv * dm);
  r[idx] = rv; z[idx] = zv; o[idx] = ov; xr[idx] = xrv; xz[idx] = xzv;
  rc[idx] = rv * cv; xrm[idx] = xrv * mv;
}

__global__ void gate2_kernel(const float* __restrict__ z, const float* __restrict__ xz,
                             float* __restrict__ c_io, const float* __restrict__ m_in,
                             float* __restrict__ m_out,
                             float* __restrict__ gc_io, float* __restrict__ gm_io) {
  int idx = blockIdx.x * 256 + threadIdx.x;
  if (idx >= 4 * 64 * HWPIX) return;
  float gcv = tanhf(gc_io[idx]);
  float gmv = tanhf(gm_io[idx]);
  float zv = z[idx], xzv = xz[idx];
  c_io[idx]  = zv * c_io[idx] + (1.f - zv) * gcv;
  m_out[idx] = xzv * m_in[idx] + (1.f - xzv) * gmv;
  gc_io[idx] = gcv;
  gm_io[idx] = gmv;
}

__global__ void gate3_kernel(const float* __restrict__ oconv, const float* __restrict__ o,
                             float* __restrict__ h, _Float16* __restrict__ hs16, int t) {
  int idx = blockIdx.x * 256 + threadIdx.x;
  if (idx >= 4 * 64 * HWPIX) return;
  float hv = o[idx] * tanhf(oconv[idx]);
  h[idx] = hv;
  if (hs16) {
    int b = idx >> 18;
    int rest = idx & ((1 << 18) - 1);
    hs16[((size_t)(b * 8 + t) << 18) + rest] = (_Float16)hv;
  }
}

__global__ void qk_build_kernel(const float* __restrict__ r, const float* __restrict__ xr,
                                const float* __restrict__ c, const float* __restrict__ m,
                                const float* __restrict__ gc, const float* __restrict__ gm,
                                _Float16* __restrict__ q16, _Float16* __restrict__ k16) {
  int idx = blockIdx.x * 256 + threadIdx.x;
  if (idx >= 4 * 64 * HWPIX) return;
  float rs = r[idx] + xr[idx];
  q16[idx] = (_Float16)((rs + c[idx] + gc[idx]) * 0.25f);
  k16[idx] = (_Float16)((rs + m[idx] + gm[idx]) * 0.25f);
}

// ---------------------------------------------------------------------------
// Attention logits: per batch, logits[64x64] = (q[64x4096] @ k^T) * scale.
// grid(4); 8 waves cover 4x4 tiles of 16x16 (each wave: 1 M-tile x 2 N-tiles).
// ---------------------------------------------------------------------------
__global__ __launch_bounds__(256) void qk_gemm_wmma(
    const _Float16* __restrict__ q16, const _Float16* __restrict__ k16,
    float* __restrict__ logits, float scale) {
  const int bb = blockIdx.x;
  const int tid = threadIdx.x, wv = tid >> 5, lane = tid & 31;
  const int laneM = lane & 15, laneK = (lane >> 4) << 3;
  const int mt  = wv & 3;
  const int ntb = (wv >> 2) * 2;
  const _Float16* qrow  = q16 + ((size_t)bb * 64 + mt * 16 + laneM) * HWPIX;
  const _Float16* krow0 = k16 + ((size_t)bb * 64 + (ntb + 0) * 16 + laneM) * HWPIX;
  const _Float16* krow1 = k16 + ((size_t)bb * 64 + (ntb + 1) * 16 + laneM) * HWPIX;
  v8f acc0 = {}, acc1 = {};
  for (int k0 = 0; k0 < HWPIX; k0 += 32) {
    v16h a  = mk16(*(const v8h*)(qrow  + k0 + laneK), *(const v8h*)(qrow  + k0 + laneK + 16));
    v16h f0 = mk16(*(const v8h*)(krow0 + k0 + laneK), *(const v8h*)(krow0 + k0 + laneK + 16));
    v16h f1 = mk16(*(const v8h*)(krow1 + k0 + laneK), *(const v8h*)(krow1 + k0 + laneK + 16));
    acc0 = __builtin_amdgcn_wmma_f32_16x16x32_f16(false, a, false, f0, (short)0, acc0, false, false);
    acc1 = __builtin_amdgcn_wmma_f32_16x16x32_f16(false, a, false, f1, (short)0, acc1, false, false);
  }
  const int mh = (lane >> 4) * 8;
#pragma unroll
  for (int r = 0; r < 8; ++r) {
    int mrow = mt * 16 + r + mh;
    logits[((size_t)bb * 64 + mrow) * 64 + (ntb + 0) * 16 + laneM] = acc0[r] * scale;
    logits[((size_t)bb * 64 + mrow) * 64 + (ntb + 1) * 16 + laneM] = acc1[r] * scale;
  }
}

__global__ void softmax64_kernel(const float* __restrict__ logits,
                                 _Float16* __restrict__ attn16) {
  int row = blockIdx.x * blockDim.x + threadIdx.x;
  if (row >= 4 * 64) return;
  const float* rp = logits + (size_t)row * 64;
  float mx = rp[0];
  for (int i = 1; i < 64; ++i) mx = fmaxf(mx, rp[i]);
  float s = 0.f;
  for (int i = 0; i < 64; ++i) s += __expf(rp[i] - mx);
  float inv = 1.f / s;
  _Float16* op = attn16 + (size_t)row * 64;
  for (int i = 0; i < 64; ++i) op[i] = (_Float16)(__expf(rp[i] - mx) * inv);
}

// ---------------------------------------------------------------------------
// out[b,t,c,n] = sum_d attn[b,c,d] * v[b,t,d,n];  M=64, K=64, N=4096.
// grid(32 ntile, 8 t, 4 b); wave w: one 16-wide N strip x 4 M-tiles.
// ---------------------------------------------------------------------------
__global__ __launch_bounds__(256) void av_gemm_wmma(
    const _Float16* __restrict__ attn16, const _Float16* __restrict__ hs16,
    float* __restrict__ out) {
  const int bb = blockIdx.z, t = blockIdx.y;
  const int tid = threadIdx.x, wv = tid >> 5, lane = tid & 31;
  const int n0 = blockIdx.x * 128 + wv * 16;
  const int laneM = lane & 15, laneK = (lane >> 4) << 3;
  const _Float16* vbase = hs16 + ((size_t)(bb * 8 + t) << 18);  // [64][4096]
  v8f acc[4] = {};
  for (int k0 = 0; k0 < 64; k0 += 32) {
    v16h bf;
#pragma unroll
    for (int e = 0; e < 8; ++e) {
      bf[e]     = vbase[(size_t)(k0 + laneK + e) * HWPIX + n0 + laneM];
      bf[e + 8] = vbase[(size_t)(k0 + laneK + 16 + e) * HWPIX + n0 + laneM];
    }
#pragma unroll
    for (int mi = 0; mi < 4; ++mi) {
      const _Float16* arow = attn16 + ((size_t)bb * 64 + mi * 16 + laneM) * 64;
      v16h af = mk16(*(const v8h*)(arow + k0 + laneK), *(const v8h*)(arow + k0 + laneK + 16));
      acc[mi] = __builtin_amdgcn_wmma_f32_16x16x32_f16(false, af, false, bf, (short)0,
                                                       acc[mi], false, false);
    }
  }
  const int mh = (lane >> 4) * 8;
#pragma unroll
  for (int mi = 0; mi < 4; ++mi)
#pragma unroll
    for (int r = 0; r < 8; ++r) {
      int mrow = mi * 16 + r + mh;
      out[((size_t)(bb * 8 + t) * 64 + mrow) * HWPIX + n0 + laneM] = acc[mi][r];
    }
}

// ---------------------------------------------------------------------------
// Host orchestration
// ---------------------------------------------------------------------------
extern "C" void kernel_launch(void* const* d_in, const int* in_sizes, int n_in,
                              void* d_out, int out_size, void* d_ws, size_t ws_size,
                              hipStream_t stream) {
  const float* x = (const float*)d_in[0];
  const float* w_rzo[2] = {(const float*)d_in[1],  (const float*)d_in[12]};
  const float* b_rzo[2] = {(const float*)d_in[2],  (const float*)d_in[13]};
  const float* w_rz [2] = {(const float*)d_in[3],  (const float*)d_in[14]};
  const float* b_rz [2] = {(const float*)d_in[4],  (const float*)d_in[15]};
  const float* w_h  [2] = {(const float*)d_in[5],  (const float*)d_in[16]};
  const float* b_h  [2] = {(const float*)d_in[6],  (const float*)d_in[17]};
  const float* w_o  [2] = {(const float*)d_in[7],  (const float*)d_in[18]};
  const float* b_o  [2] = {(const float*)d_in[8],  (const float*)d_in[19]};
  const float* td_h [2] = {(const float*)d_in[9],  (const float*)d_in[20]};
  const float* td_m [2] = {(const float*)d_in[10], (const float*)d_in[21]};
  const float* tg   [2] = {(const float*)d_in[11], (const float*)d_in[22]};

  char* wp = (char*)d_ws;
  auto alloc = [&](size_t bytes) -> void* {
    void* r = (void*)wp;
    wp += (bytes + 255) & ~(size_t)255;
    return r;
  };

  const size_t S = (size_t)4 * 64 * HWPIX;       // per-state element count
  const int Kc[2]  = {80 * 9, 128 * 9};          // real K (720, 1152)
  const int Kp[2]  = {736, 1152};                // K padded to multiple of 32
  const int KpO    = 128;                        // 1x1 conv K (already aligned)

  _Float16 *wA16[2], *wB16[2], *wH16[2], *wO16[2];
  for (int l = 0; l < 2; ++l) {
    wA16[l] = (_Float16*)alloc(sizeof(_Float16) * 192 * Kp[l]);
    wB16[l] = (_Float16*)alloc(sizeof(_Float16) * 128 * Kp[l]);
    wH16[l] = (_Float16*)alloc(sizeof(_Float16) * 64 * Kp[l]);
    wO16[l] = (_Float16*)alloc(sizeof(_Float16) * 64 * KpO);
  }
  float* stbase = (float*)alloc(sizeof(float) * S * 6);
  float* hst[2] = {stbase, stbase + 3 * S};
  float* cst[2] = {stbase + S, stbase + 4 * S};
  float* mst[2] = {stbase + 2 * S, stbase + 5 * S};

  float* Abuf = (float*)alloc(sizeof(float) * (size_t)4 * 192 * HWPIX);
  float* Bbuf = (float*)alloc(sizeof(float) * (size_t)4 * 128 * HWPIX);
  float* rb   = (float*)alloc(sizeof(float) * S);
  float* zb   = (float*)alloc(sizeof(float) * S);
  float* ob   = (float*)alloc(sizeof(float) * S);
  float* xrb  = (float*)alloc(sizeof(float) * S);
  float* xzb  = (float*)alloc(sizeof(float) * S);
  float* rcb  = (float*)alloc(sizeof(float) * S);
  float* xrmb = (float*)alloc(sizeof(float) * S);
  float* gcb  = (float*)alloc(sizeof(float) * S);
  float* gmb  = (float*)alloc(sizeof(float) * S);
  float* ocb  = (float*)alloc(sizeof(float) * S);
  _Float16* q16buf = (_Float16*)alloc(sizeof(_Float16) * S);
  _Float16* k16buf = (_Float16*)alloc(sizeof(_Float16) * S);
  _Float16* hs16   = (_Float16*)alloc(sizeof(_Float16) * S * 8);  // [B,T,64,HW]
  float*    logits = (float*)alloc(sizeof(float) * 4 * 64 * 64);
  _Float16* attn16 = (_Float16*)alloc(sizeof(_Float16) * 4 * 64 * 64);

  // zero-init recurrent states (graph-capturable)
  hipMemsetAsync(stbase, 0, sizeof(float) * S * 6, stream);

  // convert + K-pad weights fp32 -> f16
  auto cvt = [&](const float* src, _Float16* dst, int Cout, int K, int Kpad) {
    int n = Cout * Kpad;
    cvt_pad_kernel<<<dim3((n + 255) / 256), dim3(256), 0, stream>>>(src, dst, Cout, K, Kpad);
  };
  for (int l = 0; l < 2; ++l) {
    cvt(w_rzo[l], wA16[l], 192, Kc[l], Kp[l]);
    cvt(w_rz[l],  wB16[l], 128, Kc[l], Kp[l]);
    cvt(w_h[l],   wH16[l],  64, Kc[l], Kp[l]);
    cvt(w_o[l],   wO16[l],  64, 128,   KpO);
  }

  const int EW_BLOCKS = (int)((S + 255) / 256);

  // templated conv dispatch: (layer, 3x3|1x1)
  auto conv3_l0 = [&](const _Float16* Wt, const float* bias,
                      const float* s0, long long bs0,
                      const float* s1, long long bs1, float* outb, int Cout) {
    dim3 grid(32, Cout / 64, 4);
    conv_gemm_wmma<16, 64, 9, 1><<<grid, dim3(256), 0, stream>>>(
        Wt, bias, s0, bs0, s1, bs1, outb, Cout, Kp[0]);
  };
  auto conv3_l1 = [&](const _Float16* Wt, const float* bias,
                      const float* s0, long long bs0,
                      const float* s1, long long bs1, float* outb, int Cout) {
    dim3 grid(32, Cout / 64, 4);
    conv_gemm_wmma<64, 64, 9, 1><<<grid, dim3(256), 0, stream>>>(
        Wt, bias, s0, bs0, s1, bs1, outb, Cout, Kp[1]);
  };
  auto conv1 = [&](const _Float16* Wt, const float* bias,
                   const float* s0, long long bs0,
                   const float* s1, long long bs1, float* outb, int Cout) {
    dim3 grid(32, Cout / 64, 4);
    conv_gemm_wmma<64, 64, 1, 0><<<grid, dim3(256), 0, stream>>>(
        Wt, bias, s0, bs0, s1, bs1, outb, Cout, KpO);
  };

  for (int t = 0; t < 8; ++t) {
    for (int l = 0; l < 2; ++l) {
      const float* xin; long long xbs;
      if (l == 0) { xin = x + (size_t)t * 16 * HWPIX; xbs = 8LL * 16 * HWPIX; }
      else        { xin = hst[0];                      xbs = 64LL * HWPIX;     }
      const float* minp = (l == 0) ? mst[1] : mst[0];
      const long long sbs = 64LL * HWPIX;

      if (l == 0) {
        conv3_l0(wA16[l], b_rzo[l], xin, xbs, hst[l], sbs, Abuf, 192);
        conv3_l0(wB16[l], b_rz[l],  xin, xbs, minp,   sbs, Bbuf, 128);
      } else {
        conv3_l1(wA16[l], b_rzo[l], xin, xbs, hst[l], sbs, Abuf, 192);
        conv3_l1(wB16[l], b_rz[l],  xin, xbs, minp,   sbs, Bbuf, 128);
      }
      gate1_kernel<<<EW_BLOCKS, 256, 0, stream>>>(Abuf, Bbuf, hst[l], cst[l], minp,
                                                  td_h[l], td_m[l], tg[l],
                                                  rb, zb, ob, xrb, xzb, rcb, xrmb);
      if (l == 0) {
        conv3_l0(wH16[l], b_h[l], xin, xbs, rcb,  sbs, gcb, 64);
        conv3_l0(wH16[l], b_h[l], xin, xbs, xrmb, sbs, gmb, 64);
      } else {
        conv3_l1(wH16[l], b_h[l], xin, xbs, rcb,  sbs, gcb, 64);
        conv3_l1(wH16[l], b_h[l], xin, xbs, xrmb, sbs, gmb, 64);
      }
      gate2_kernel<<<EW_BLOCKS, 256, 0, stream>>>(zb, xzb, cst[l], minp, mst[l], gcb, gmb);
      conv1(wO16[l], b_o[l], cst[l], sbs, mst[l], sbs, ocb, 64);
      gate3_kernel<<<EW_BLOCKS, 256, 0, stream>>>(ocb, ob, hst[l],
                                                  (l == 1) ? hs16 : (_Float16*)nullptr, t);
    }
  }

  // attention over channels (gates are the last layer / last step values)
  qk_build_kernel<<<EW_BLOCKS, 256, 0, stream>>>(rb, xrb, cst[1], mst[1], gcb, gmb,
                                                 q16buf, k16buf);
  qk_gemm_wmma<<<dim3(4), dim3(256), 0, stream>>>(q16buf, k16buf, logits,
                                                  0.015625f /* 1/sqrt(4096) */);
  softmax64_kernel<<<dim3(1), dim3(256), 0, stream>>>(logits, attn16);
  av_gemm_wmma<<<dim3(32, 8, 4), dim3(256), 0, stream>>>(attn16, hs16, (float*)d_out);
}
